// MLA_59889023975547
// MI455X (gfx1250) — compile-verified, matrix operand order checked by reading, not run on hardware
//
#include <hip/hip_runtime.h>
#include <hip/hip_bf16.h>

typedef __attribute__((ext_vector_type(16))) __bf16 v16bf;
typedef __attribute__((ext_vector_type(8)))  float  v8f;
typedef __attribute__((ext_vector_type(4)))  int    v4i;

// ---------- CDNA5 async global->LDS copy (B128), guarded ----------
#if defined(__has_builtin)
#  if __has_builtin(__builtin_amdgcn_global_load_async_to_lds_b128)
#    define USE_ASYNC_LDS 1
#  else
#    define USE_ASYNC_LDS 0
#  endif
#else
#  define USE_ASYNC_LDS 0
#endif

#if USE_ASYNC_LDS
typedef __attribute__((address_space(1))) v4i* gv4i_p;   // global int4*
typedef __attribute__((address_space(3))) v4i* lv4i_p;   // LDS int4*

__device__ __forceinline__ void async_cp16(unsigned short* lds,
                                           const unsigned short* g) {
    __builtin_amdgcn_global_load_async_to_lds_b128(
        (gv4i_p)(g), (lv4i_p)(lds), 0, 0);
}
#endif

__device__ __forceinline__ void wait_async0() {
#if defined(__has_builtin) && __has_builtin(__builtin_amdgcn_s_wait_asynccnt)
    __builtin_amdgcn_s_wait_asynccnt(0);
#else
    asm volatile("s_wait_asynccnt 0" ::: "memory");
#endif
}

// ---------- helpers ----------
__device__ __forceinline__ unsigned short f2bf(float f) {
    unsigned int u = __float_as_uint(f);
    u += 0x7FFFu + ((u >> 16) & 1u);        // round-to-nearest-even
    return (unsigned short)(u >> 16);
}

// Load one 16-bit WMMA A/B fragment (16x32 tile slice) from a ushort buffer.
// Per CDNA5 ISA layout: lane holds row (lane&15); VGPR v holds K pair at
//   v<4 : k = 2v + 8*half        v>=4 : k = 16 + 2(v-4) + 8*half
__device__ __forceinline__ v16bf frag_ld(const unsigned short* __restrict__ p,
                                         int stride, int row, int kb, int half) {
    union { v16bf v; unsigned int u[8]; } f;
#pragma unroll
    for (int r = 0; r < 8; ++r) {
        int k = kb + ((r < 4) ? (2 * r + half * 8) : (16 + 2 * (r - 4) + half * 8));
        f.u[r] = *(const unsigned int*)(p + (size_t)row * stride + k);
    }
    return f.v;
}

// ---------- f32 -> bf16 convert ----------
__global__ void conv_f32_bf16(const float* __restrict__ s,
                              unsigned short* __restrict__ d, size_t n) {
    size_t i  = (size_t)blockIdx.x * blockDim.x + threadIdx.x;
    size_t st = (size_t)gridDim.x * blockDim.x;
    for (; i < n; i += st) d[i] = f2bf(s[i]);
}

// ---------- generic WMMA GEMM:  C[M,N] = A[M,K] * W[N,K]^T + bias ----------
// mode 0: DyT epilogue -> bf16 out    (out = gamma*tanh(alpha*y)+beta)
// mode 1: bias only    -> bf16 out
// mode 2: bias only    -> f32 out
#define GBM 128
#define GBN 128
#define GBK 32
#define GLD 40   // padded LDS row stride (ushorts)

__global__ __launch_bounds__(256)
void gemm_wmma(const unsigned short* __restrict__ A,
               const unsigned short* __restrict__ W,
               const float* __restrict__ bias,
               const float* __restrict__ alpha_p,
               const float* __restrict__ gamma,
               const float* __restrict__ beta,
               unsigned short* __restrict__ Obf,
               float* __restrict__ Of,
               int M, int N, int K, int mode) {
    __shared__ unsigned short sA[2][GBM * GLD];
    __shared__ unsigned short sB[2][GBN * GLD];

    const int tid  = threadIdx.x;
    const int lane = tid & 31;
    const int wave = tid >> 5;      // 0..7
    const int wm   = wave >> 1;     // 0..3  (32-row strip)
    const int wn   = wave & 1;      // 0..1  (64-col strip)
    const int m0   = blockIdx.y * GBM;
    const int n0   = blockIdx.x * GBN;
    const int row  = lane & 15;
    const int half = lane >> 4;

    v8f acc[2][4];
#pragma unroll
    for (int i = 0; i < 2; ++i)
#pragma unroll
        for (int j = 0; j < 4; ++j) acc[i][j] = (v8f){0, 0, 0, 0, 0, 0, 0, 0};

    const int ldr = tid >> 1;           // 0..127 tile row
    const int ldc = (tid & 1) * 16;     // col base 0/16

    const unsigned short* gA = A + (size_t)(m0 + ldr) * K + ldc;
    const unsigned short* gW = W + (size_t)(n0 + ldr) * K + ldc;
    const int lofs = ldr * GLD + ldc;

    // stage one 32-wide K tile pair into LDS buffer `buf`
    auto stage = [&](int buf, int kt) {
#if USE_ASYNC_LDS
        async_cp16(&sA[buf][lofs],     gA + kt);
        async_cp16(&sA[buf][lofs + 8], gA + kt + 8);
        async_cp16(&sB[buf][lofs],     gW + kt);
        async_cp16(&sB[buf][lofs + 8], gW + kt + 8);
#else
        const uint4* ga = (const uint4*)(gA + kt);
        const uint4* gw = (const uint4*)(gW + kt);
        uint4 a0 = ga[0], a1 = ga[1];
        uint4 w0 = gw[0], w1 = gw[1];
        *(uint4*)(&sA[buf][lofs])     = a0;
        *(uint4*)(&sA[buf][lofs + 8]) = a1;
        *(uint4*)(&sB[buf][lofs])     = w0;
        *(uint4*)(&sB[buf][lofs + 8]) = w1;
#endif
    };

    const int niter = K / GBK;
    stage(0, 0);
    for (int it = 0; it < niter; ++it) {
        const int cur = it & 1;
#if USE_ASYNC_LDS
        wait_async0();                 // my copies into `cur` are done
#endif
        __syncthreads();               // everyone's copies done; prev reads done
        if (it + 1 < niter) stage(cur ^ 1, (it + 1) * GBK);

        v16bf af[2], bfr[4];
#pragma unroll
        for (int i = 0; i < 2; ++i)
            af[i] = frag_ld(&sA[cur][(wm * 32 + i * 16) * GLD], GLD, row, 0, half);
#pragma unroll
        for (int j = 0; j < 4; ++j)
            bfr[j] = frag_ld(&sB[cur][(wn * 64 + j * 16) * GLD], GLD, row, 0, half);
#pragma unroll
        for (int i = 0; i < 2; ++i)
#pragma unroll
            for (int j = 0; j < 4; ++j)
                acc[i][j] = __builtin_amdgcn_wmma_f32_16x16x32_bf16(
                    false, af[i], false, bfr[j], (short)0, acc[i][j], false, false);
    }

    // epilogue
#pragma unroll
    for (int j = 0; j < 4; ++j) {
        const int gn = n0 + wn * 64 + j * 16 + row;
        const float b = bias[gn];
        float ga = 0.f, be = 0.f, al = 0.f;
        if (mode == 0) { ga = gamma[gn]; be = beta[gn]; al = alpha_p[0]; }
#pragma unroll
        for (int i = 0; i < 2; ++i) {
#pragma unroll
            for (int r = 0; r < 8; ++r) {
                const int gm = m0 + wm * 32 + i * 16 + r + half * 8;
                float y = acc[i][j][r] + b;
                if (mode == 0) {
                    y = ga * tanhf(al * y) + be;
                    Obf[(size_t)gm * N + gn] = f2bf(y);
                } else if (mode == 1) {
                    Obf[(size_t)gm * N + gn] = f2bf(y);
                } else {
                    Of[(size_t)gm * N + gn] = y;
                }
            }
        }
    }
}

// ---------- flash attention (causal), bf16 in / bf16 out ----------
#define AQT 64
#define AKT 64
#define KST 136   // sK stride (ushort), 64 rows x 128 dims
#define VST 66    // sVT stride (ushort), 128 dims x 64 keys
#define SST 66    // sS stride (float),  64 x 64 scores
#define PST 68    // sP stride (ushort), 64 x 64 probs

__global__ __launch_bounds__(128)
void mla_attn(const unsigned short* __restrict__ Q,   // (B*S, H*128)
              const unsigned short* __restrict__ KV,  // (B*S, H*256), [k|v]
              unsigned short* __restrict__ O,         // (B*S, H*128)
              int Bc, int Sc, int Hc) {
    __shared__ unsigned short sK[AKT * KST];
    __shared__ unsigned short sVT[128 * VST];
    __shared__ float          sS[AQT * SST];
    __shared__ unsigned short sP[AQT * PST];
    __shared__ float          sScale[AQT];
    __shared__ float          sL[AQT];

    const int tid   = threadIdx.x;
    const int lane  = tid & 31;
    const int wave  = tid >> 5;          // 0..3, owns 16 query rows
    const int bh    = blockIdx.x;
    const int b     = bh / Hc, h = bh % Hc;
    const int qb    = blockIdx.y;
    const int qBase = qb * AQT;
    const int row   = lane & 15;
    const int half  = lane >> 4;
    const int DQ    = Hc * 128;          // q row stride
    const int DKV   = Hc * 256;          // kv row stride
    const float scale = 0.08838834764831845f;  // 128^-0.5

    // Q fragments for this wave's 16 rows (kept in registers)
    const unsigned short* qp = Q + (size_t)(b * Sc + qBase + wave * 16) * DQ + h * 128;
    v16bf qf[4];
#pragma unroll
    for (int kf = 0; kf < 4; ++kf) qf[kf] = frag_ld(qp, DQ, row, kf * 32, half);

    v8f of[8];
#pragma unroll
    for (int f = 0; f < 8; ++f) of[f] = (v8f){0, 0, 0, 0, 0, 0, 0, 0};
    float m_i = -3.0e38f, l_i = 0.0f;    // used by tid < 64 only

    const int nkb = qb + 1;              // causal: only blocks <= diag
    for (int kb = 0; kb < nkb; ++kb) {
        __syncthreads();                 // prev iter done reading sK/sVT
        {   // stage K tile (async) + transposed V tile (through VGPRs)
            const int r  = tid >> 1;            // key row 0..63
            const int c0 = (tid & 1) * 64;      // dim half
            const unsigned short* kp =
                KV + (size_t)(b * Sc + kb * AKT + r) * DKV + h * 256;
#if USE_ASYNC_LDS
#pragma unroll
            for (int q8 = 0; q8 < 8; ++q8)
                async_cp16(sK + r * KST + c0 + q8 * 8, kp + c0 + q8 * 8);
#else
            const uint4* gk = (const uint4*)(kp + c0);
#pragma unroll
            for (int q8 = 0; q8 < 8; ++q8)
                *(uint4*)(sK + r * KST + c0 + q8 * 8) = gk[q8];
#endif
            const unsigned short* vp = kp + 128;
#pragma unroll
            for (int e = 0; e < 64; ++e)
                sVT[(size_t)(c0 + e) * VST + r] = vp[c0 + e];
#if USE_ASYNC_LDS
            wait_async0();
#endif
        }
        __syncthreads();

        // S = Q K^T  (each wave: 16 x 64 strip)
#pragma unroll
        for (int nf = 0; nf < 4; ++nf) {
            v8f s = (v8f){0, 0, 0, 0, 0, 0, 0, 0};
#pragma unroll
            for (int ks = 0; ks < 4; ++ks) {
                v16bf kfb = frag_ld(sK + nf * 16 * KST, KST, row, ks * 32, half);
                s = __builtin_amdgcn_wmma_f32_16x16x32_bf16(
                    false, qf[ks], false, kfb, (short)0, s, false, false);
            }
            const int gq0 = qBase + wave * 16 + half * 8;
            const int gk0 = kb * AKT + nf * 16 + row;
#pragma unroll
            for (int r8 = 0; r8 < 8; ++r8) {
                float v = s[r8] * scale;
                if (gk0 > gq0 + r8) v = -1.0e9f;   // causal mask (== NEG add)
                sS[(wave * 16 + r8 + half * 8) * SST + nf * 16 + row] = v;
            }
        }
        __syncthreads();

        // online softmax stats: thread t handles query row t
        if (tid < AQT) {
            float mx = m_i;
            for (int c = 0; c < AKT; ++c) {
                float v = sS[tid * SST + c];
                mx = v > mx ? v : mx;
            }
            float corr = __expf(m_i - mx);
            float lsum = l_i * corr;
            for (int c = 0; c < AKT; ++c) {
                float p = __expf(sS[tid * SST + c] - mx);
                lsum += p;
                sP[tid * PST + c] = f2bf(p);
            }
            sScale[tid] = corr;
            m_i = mx;
            l_i = lsum;
        }
        __syncthreads();

        // rescale O, then O += P V
        float sc[8];
#pragma unroll
        for (int r8 = 0; r8 < 8; ++r8) sc[r8] = sScale[wave * 16 + r8 + half * 8];
#pragma unroll
        for (int f = 0; f < 8; ++f)
#pragma unroll
            for (int r8 = 0; r8 < 8; ++r8) of[f][r8] *= sc[r8];

        v16bf pf[2];
#pragma unroll
        for (int ks = 0; ks < 2; ++ks)
            pf[ks] = frag_ld(sP + wave * 16 * PST, PST, row, ks * 32, half);
#pragma unroll
        for (int nf = 0; nf < 8; ++nf) {
#pragma unroll
            for (int ks = 0; ks < 2; ++ks) {
                v16bf vf = frag_ld(sVT + nf * 16 * VST, VST, row, ks * 32, half);
                of[nf] = __builtin_amdgcn_wmma_f32_16x16x32_bf16(
                    false, pf[ks], false, vf, (short)0, of[nf], false, false);
            }
        }
    }

    if (tid < AQT) sL[tid] = l_i;
    __syncthreads();

    float inv[8];
#pragma unroll
    for (int r8 = 0; r8 < 8; ++r8)
        inv[r8] = 1.0f / sL[wave * 16 + r8 + half * 8];

    unsigned short* op = O + (size_t)(b * Sc + qBase + wave * 16) * DQ + h * 128;
#pragma unroll
    for (int nf = 0; nf < 8; ++nf)
#pragma unroll
        for (int r8 = 0; r8 < 8; ++r8)
            op[(size_t)(r8 + half * 8) * DQ + nf * 16 + row] =
                f2bf(of[nf][r8] * inv[r8]);
}

// ---------- host launcher ----------
extern "C" void kernel_launch(void* const* d_in, const int* in_sizes, int n_in,
                              void* d_out, int out_size, void* d_ws, size_t ws_size,
                              hipStream_t stream) {
    const float* x        = (const float*)d_in[0];
    const float* wq_a_w   = (const float*)d_in[3];
    const float* wq_a_b   = (const float*)d_in[4];
    const float* q_alpha  = (const float*)d_in[5];
    const float* q_gamma  = (const float*)d_in[6];
    const float* q_beta   = (const float*)d_in[7];
    const float* wq_b_w   = (const float*)d_in[8];
    const float* wq_b_b   = (const float*)d_in[9];
    const float* wkv_a_w  = (const float*)d_in[10];
    const float* wkv_a_b  = (const float*)d_in[11];
    const float* kv_alpha = (const float*)d_in[12];
    const float* kv_gamma = (const float*)d_in[13];
    const float* kv_beta  = (const float*)d_in[14];
    const float* wkv_b_w  = (const float*)d_in[15];
    const float* wkv_b_b  = (const float*)d_in[16];
    const float* wo_w     = (const float*)d_in[17];
    const float* wo_b     = (const float*)d_in[18];
    float* out = (float*)d_out;

    const int B = 4, S = 2048, DIM = 2048, H = 16, QR = 1024, KVR = 512;
    const int M = B * S;

    char* ws = (char*)d_ws;
    size_t off = 0;
    auto alloc = [&](size_t elems) -> unsigned short* {
        unsigned short* p = (unsigned short*)(ws + off);
        off += ((elems * 2 + 255) & ~(size_t)255);
        return p;
    };
    unsigned short* x_bf    = alloc((size_t)M * DIM);
    unsigned short* wqa_bf  = alloc((size_t)QR * DIM);
    unsigned short* wqb_bf  = alloc((size_t)(H * 128) * QR);
    unsigned short* wkva_bf = alloc((size_t)KVR * DIM);
    unsigned short* wkvb_bf = alloc((size_t)(H * 256) * KVR);
    unsigned short* wo_bf   = alloc((size_t)DIM * (H * 128));
    unsigned short* qa_bf   = alloc((size_t)M * QR);
    unsigned short* kva_bf  = alloc((size_t)M * KVR);
    unsigned short* q_bf    = alloc((size_t)M * (H * 128));
    unsigned short* kv_bf   = alloc((size_t)M * (H * 256));
    unsigned short* at_bf   = alloc((size_t)M * (H * 128));

    auto conv = [&](const float* s, unsigned short* d, size_t n) {
        int blocks = (int)((n + 255) / 256);
        if (blocks > 4096) blocks = 4096;
        conv_f32_bf16<<<blocks, 256, 0, stream>>>(s, d, n);
    };
    conv(x,       x_bf,    (size_t)M * DIM);
    conv(wq_a_w,  wqa_bf,  (size_t)QR * DIM);
    conv(wq_b_w,  wqb_bf,  (size_t)(H * 128) * QR);
    conv(wkv_a_w, wkva_bf, (size_t)KVR * DIM);
    conv(wkv_b_w, wkvb_bf, (size_t)(H * 256) * KVR);
    conv(wo_w,    wo_bf,   (size_t)DIM * (H * 128));

    auto gemm = [&](const unsigned short* A, const unsigned short* W,
                    const float* bias, const float* al, const float* ga,
                    const float* be, unsigned short* Ob, float* Ofp,
                    int m, int n, int k, int mode) {
        dim3 grid(n / GBN, m / GBM);
        gemm_wmma<<<grid, 256, 0, stream>>>(A, W, bias, al, ga, be, Ob, Ofp,
                                            m, n, k, mode);
    };

    // q_a = DyT(x @ wq_a^T + b)
    gemm(x_bf, wqa_bf, wq_a_b, q_alpha, q_gamma, q_beta, qa_bf, nullptr,
         M, QR, DIM, 0);
    // kv_a = DyT(x @ wkv_a^T + b)
    gemm(x_bf, wkva_bf, wkv_a_b, kv_alpha, kv_gamma, kv_beta, kva_bf, nullptr,
         M, KVR, DIM, 0);
    // q = q_a @ wq_b^T + b
    gemm(qa_bf, wqb_bf, wq_b_b, nullptr, nullptr, nullptr, q_bf, nullptr,
         M, H * 128, QR, 1);
    // kv = kv_a @ wkv_b^T + b
    gemm(kva_bf, wkvb_bf, wkv_b_b, nullptr, nullptr, nullptr, kv_bf, nullptr,
         M, H * 256, KVR, 1);
    // attention
    dim3 agrid(B * H, S / AQT);
    mla_attn<<<agrid, 128, 0, stream>>>(q_bf, kv_bf, at_bf, B, S, H);
    // out = attn @ wo^T + b   (f32 out)
    gemm(at_bf, wo_bf, wo_b, nullptr, nullptr, nullptr, nullptr, out,
         M, DIM, H * 128, 2);
}
